// GatedGraphClassifier_14362370638538
// MI455X (gfx1250) — matrix operand order repaired and checked by reading, not compiled, not run on hardware
//
#include <hip/hip_runtime.h>
#include <hip/hip_bf16.h>

// ---------------------------------------------------------------------------
// GatedGraphConv classifier for MI455X (gfx1250, wave32, WMMA bf16)
// ---------------------------------------------------------------------------

typedef __attribute__((ext_vector_type(16))) __bf16 v16bf;
typedef __attribute__((ext_vector_type(8)))  float  v8f;

#define HDIM   64
#define XDIM   79
#define NBLK   4
#define NLAY   2

__device__ __forceinline__ __bf16 f2bf(float f) { return static_cast<__bf16>(f); }

// ISA 16-bit A-matrix 16x32 layout: lanes 0-15: K={0..7,16..23}; lanes 16-31: +8
__device__ __forceinline__ int kmapA(int lane, int p) {
    return ((p < 8) ? p : p + 8) + ((lane >> 4) << 3);
}
// ISA 16-bit B-matrix 32x16 layout: lanes 0-15: K={0..7,16..23}; lanes 16-31: +8
__device__ __forceinline__ int kmapB(int lane, int p) {
    return ((p < 8) ? p : p + 8) + ((lane >> 4) << 3);
}

// Scalar (bounds-checked) A-fragment gather; used for K=79 projection only.
__device__ __forceinline__ v16bf load_a_frag_any(const float* __restrict__ A, long rowbase,
                                                 int ldA, int K, int kstep, int lane, int Nrows) {
    long row = rowbase + (lane & 15);
    if (row >= Nrows) row = Nrows - 1;           // keep EXEC full
    const float* rp = A + row * (long)ldA;
    v16bf a;
#pragma unroll
    for (int p = 0; p < 16; ++p) {
        int k = kstep * 32 + kmapA(lane, p);
        a[p] = (k < K) ? f2bf(rp[k]) : f2bf(0.0f);
    }
    return a;
}

// Vectorized A-fragment gather for K multiple of 32: two contiguous 8-float
// runs per lane -> 4x global_load_b128 + packed cvt.
__device__ __forceinline__ v16bf load_a_frag64(const float* __restrict__ A, long rowbase,
                                               int ldA, int kstep, int lane, int Nrows) {
    long row = rowbase + (lane & 15);
    if (row >= Nrows) row = Nrows - 1;
    const float* rp = A + row * (long)ldA + kstep * 32 + ((lane >> 4) << 3);
    float4 c0 = *(const float4*)(rp);
    float4 c1 = *(const float4*)(rp + 4);
    float4 c2 = *(const float4*)(rp + 16);
    float4 c3 = *(const float4*)(rp + 20);
    v16bf a;
    a[0]  = f2bf(c0.x); a[1]  = f2bf(c0.y); a[2]  = f2bf(c0.z); a[3]  = f2bf(c0.w);
    a[4]  = f2bf(c1.x); a[5]  = f2bf(c1.y); a[6]  = f2bf(c1.z); a[7]  = f2bf(c1.w);
    a[8]  = f2bf(c2.x); a[9]  = f2bf(c2.y); a[10] = f2bf(c2.z); a[11] = f2bf(c2.w);
    a[12] = f2bf(c3.x); a[13] = f2bf(c3.y); a[14] = f2bf(c3.z); a[15] = f2bf(c3.w);
    return a;
}

// ---------------------------------------------------------------------------
// Weight repack: fp32 matrix -> bf16 WMMA B-fragment layout
// dst index == fragment load index: ((ks*NT+nt)*32+lane)*16 + p
// ---------------------------------------------------------------------------
__global__ void pack_b_frags(const float* __restrict__ src, int K, int ldsrc,
                             int NT, int KSTEPS, int transpose, __bf16* __restrict__ dst) {
    int idx = blockIdx.x * blockDim.x + threadIdx.x;
    int total = KSTEPS * NT * 512;
    if (idx >= total) return;
    int p    = idx & 15;
    int lane = (idx >> 4) & 31;
    int t    = idx >> 9;              // ks*NT + nt
    int nt   = t % NT;
    int ks   = t / NT;
    int k = ks * 32 + kmapB(lane, p);
    int n = nt * 16 + (lane & 15);
    float v = 0.0f;
    if (k < K) v = transpose ? src[(long)n * ldsrc + k] : src[(long)k * ldsrc + n];
    dst[idx] = f2bf(v);
}

__global__ void zero_f32(float* __restrict__ p, long n) {
    long i = blockIdx.x * (long)blockDim.x + threadIdx.x;
    if (i < n) p[i] = 0.0f;
}

// ---------------------------------------------------------------------------
// WMMA GEMM: C[N, NT*16] = A[N,K] (fp32->bf16) @ Bfrags (+bias).
// Optionally zeroes zbuf (same shape as C) for the downstream scatter.
// ---------------------------------------------------------------------------
template <int NT, int KSTEPS, bool ALIGNED>
__global__ __launch_bounds__(256, 1)
void gemm_bf16_wmma(const float* __restrict__ A, int ldA, int K,
                    const __bf16* __restrict__ Bfrags,
                    const float* __restrict__ bias,
                    float* __restrict__ C, int ldC,
                    float* __restrict__ zbuf,
                    int Nrows, int ntiles) {
    int tile = blockIdx.x * (blockDim.x >> 5) + (threadIdx.x >> 5);
    if (tile >= ntiles) return;                 // whole-wave exit: EXEC stays full
    int lane = threadIdx.x & 31;
    long rb  = (long)tile * 16;

    v8f acc[NT] = {};
#pragma unroll
    for (int ks = 0; ks < KSTEPS; ++ks) {
        v16bf a = ALIGNED ? load_a_frag64(A, rb, ldA, ks, lane, Nrows)
                          : load_a_frag_any(A, rb, ldA, K, ks, lane, Nrows);
#pragma unroll
        for (int nt = 0; nt < NT; ++nt) {
            v16bf b = *(const v16bf*)(Bfrags + (((long)(ks * NT + nt) * 32 + lane) << 4));
            acc[nt] = __builtin_amdgcn_wmma_f32_16x16x32_bf16(
                false, a, false, b, (short)0, acc[nt], false, false);
        }
    }
    int m0   = (lane >> 4) << 3;
    int col0 = lane & 15;
#pragma unroll
    for (int nt = 0; nt < NT; ++nt) {
        int col = nt * 16 + col0;
        float bv = bias ? bias[col] : 0.0f;
#pragma unroll
        for (int v = 0; v < 8; ++v) {
            long row = rb + m0 + v;
            if (row < Nrows) {
                C[row * (long)ldC + col] = acc[nt][v] + bv;
                if (zbuf) zbuf[row * (long)ldC + col] = 0.0f;
            }
        }
    }
}

// ---------------------------------------------------------------------------
// Edge scatter: agg[dst] += m[src].  One wave per edge, float2 per lane,
// global_atomic_add_f32 (L2 atomics; m and agg are L2-resident).
// ---------------------------------------------------------------------------
__global__ void scatter_edges(const float* __restrict__ m,
                              const int* __restrict__ srcs, const int* __restrict__ dsts,
                              float* __restrict__ agg, int E) {
    long tid = blockIdx.x * (long)blockDim.x + threadIdx.x;
    int e = (int)(tid >> 5);
    if (e >= E) return;
    int lane = (int)(tid & 31);
    int s = srcs[e], d = dsts[e];
    const float2 v = *(const float2*)(m + (long)s * HDIM + lane * 2);
    unsafeAtomicAdd(agg + (long)d * HDIM + lane * 2, v.x);
    unsafeAtomicAdd(agg + (long)d * HDIM + lane * 2 + 1, v.y);
}

// ---------------------------------------------------------------------------
// Fused GRU cell, one wave per 16-row tile, in-place h update.
// Gate order r -> n -> z keeps at most ~160 VGPRs live (r dies after n).
// 48 v_wmma_f32_16x16x32_bf16 per tile.
// ---------------------------------------------------------------------------
__global__ __launch_bounds__(256, 1)
void gru_fused(const float* __restrict__ agg, float* __restrict__ h,
               const __bf16* __restrict__ WihF, const __bf16* __restrict__ WhhF,
               const float* __restrict__ bih, const float* __restrict__ bhh,
               int Nrows, int ntiles, int relu) {
    int tile = blockIdx.x * (blockDim.x >> 5) + (threadIdx.x >> 5);
    if (tile >= ntiles) return;
    int lane = threadIdx.x & 31;
    long rb  = (long)tile * 16;
    int m0   = (lane >> 4) << 3;
    int col0 = lane & 15;

    v16bf aAgg[2], aH[2];
#pragma unroll
    for (int ks = 0; ks < 2; ++ks) {
        aAgg[ks] = load_a_frag64(agg, rb, HDIM, ks, lane, Nrows);
        aH[ks]   = load_a_frag64(h,   rb, HDIM, ks, lane, Nrows);
    }

    // helper to run one gate's 16 WMMAs (gi and gh accumulators)
    auto gate_mm = [&](int gate, v8f (&ai)[4], v8f (&ah2)[4]) {
#pragma unroll
        for (int ks = 0; ks < 2; ++ks) {
#pragma unroll
            for (int nt = 0; nt < 4; ++nt) {
                long fidx = ((long)(ks * 12 + gate * 4 + nt) * 32 + lane) << 4;
                v16bf bi = *(const v16bf*)(WihF + fidx);
                ai[nt] = __builtin_amdgcn_wmma_f32_16x16x32_bf16(
                    false, aAgg[ks], false, bi, (short)0, ai[nt], false, false);
                v16bf bh = *(const v16bf*)(WhhF + fidx);
                ah2[nt] = __builtin_amdgcn_wmma_f32_16x16x32_bf16(
                    false, aH[ks], false, bh, (short)0, ah2[nt], false, false);
            }
        }
    };

    // --- gate r (cols 0..63) ---
    v8f r[4];
    {
        v8f ai[4] = {}, ah2[4] = {};
        gate_mm(0, ai, ah2);
#pragma unroll
        for (int nt = 0; nt < 4; ++nt)
#pragma unroll
            for (int v = 0; v < 8; ++v) {
                int col = nt * 16 + col0;
                float s = ai[nt][v] + bih[col] + ah2[nt][v] + bhh[col];
                r[nt][v] = 1.0f / (1.0f + __expf(-s));
            }
    }

    // --- gate n (cols 128..191), consumes r ---
    v8f nn[4];
    {
        v8f ai[4] = {}, ah2[4] = {};
        gate_mm(2, ai, ah2);
#pragma unroll
        for (int nt = 0; nt < 4; ++nt)
#pragma unroll
            for (int v = 0; v < 8; ++v) {
                int col = 128 + nt * 16 + col0;
                float gi = ai[nt][v] + bih[col];
                float gh = ah2[nt][v] + bhh[col];
                nn[nt][v] = tanhf(gi + r[nt][v] * gh);
            }
    }

    // --- gate z (cols 64..127) + final blend ---
    {
        v8f ai[4] = {}, ah2[4] = {};
        gate_mm(1, ai, ah2);
#pragma unroll
        for (int nt = 0; nt < 4; ++nt)
#pragma unroll
            for (int v = 0; v < 8; ++v) {
                int col = 64 + nt * 16 + col0;
                float s  = ai[nt][v] + bih[col] + ah2[nt][v] + bhh[col];
                float zz = 1.0f / (1.0f + __expf(-s));
                long row = rb + m0 + v;
                if (row < Nrows) {
                    long off  = row * (long)HDIM + nt * 16 + col0;
                    float hol = h[off];
                    float hn  = (1.0f - zz) * nn[nt][v] + zz * hol;
                    if (relu) hn = hn > 0.0f ? hn : 0.0f;
                    h[off] = hn;
                }
            }
    }
}

// ---------------------------------------------------------------------------
// Mean-pool scatter: pooled[batch[n]] += h[n]; counts[batch[n]] += 1
// ---------------------------------------------------------------------------
__global__ void pool_scatter(const float* __restrict__ h, const int* __restrict__ batch,
                             float* __restrict__ pooled, float* __restrict__ counts, int Nrows) {
    long tid = blockIdx.x * (long)blockDim.x + threadIdx.x;
    int n = (int)(tid >> 5);
    if (n >= Nrows) return;
    int lane = (int)(tid & 31);
    int g = batch[n];
    const float2 v = *(const float2*)(h + (long)n * HDIM + lane * 2);
    unsafeAtomicAdd(pooled + (long)g * HDIM + lane * 2, v.x);
    unsafeAtomicAdd(pooled + (long)g * HDIM + lane * 2 + 1, v.y);
    if (lane == 0) unsafeAtomicAdd(counts + g, 1.0f);
}

// ---------------------------------------------------------------------------
// MLP head: 512 graphs, one thread each (negligible FLOPs)
// ---------------------------------------------------------------------------
__global__ void head_kernel(const float* __restrict__ pooled, const float* __restrict__ counts,
                            const float* __restrict__ W1, const float* __restrict__ b1,
                            const float* __restrict__ W2, const float* __restrict__ b2,
                            const float* __restrict__ W3, const float* __restrict__ b3,
                            float* __restrict__ out, int G) {
    int g = blockIdx.x * blockDim.x + threadIdx.x;
    if (g >= G) return;
    float c = counts[g]; c = c < 1.0f ? 1.0f : c;
    float p[64], t1[64], t2[32];
    for (int i = 0; i < 64; ++i) p[i] = pooled[(long)g * HDIM + i] / c;
    for (int j = 0; j < 64; ++j) {
        float s = b1[j];
        for (int i = 0; i < 64; ++i) s += p[i] * W1[j * 64 + i];
        t1[j] = s;
    }
    for (int j = 0; j < 32; ++j) {
        float s = b2[j];
        for (int i = 0; i < 64; ++i) s += t1[i] * W2[j * 64 + i];
        t2[j] = s;
    }
    float o = b3[0];
    for (int i = 0; i < 32; ++i) o += t2[i] * W3[i];
    out[g] = 1.0f / (1.0f + __expf(-o));
}

// ---------------------------------------------------------------------------
// Launch
// ---------------------------------------------------------------------------
extern "C" void kernel_launch(void* const* d_in, const int* in_sizes, int n_in,
                              void* d_out, int out_size, void* d_ws, size_t ws_size,
                              hipStream_t stream) {
    const float* x        = (const float*)d_in[0];
    const int*   eidx     = (const int*)d_in[1];
    const int*   batch    = (const int*)d_in[2];
    const float* W_proj   = (const float*)d_in[3];
    const float* b_proj   = (const float*)d_in[4];
    const float* ggc_W    = (const float*)d_in[5];
    const float* gru_Wih  = (const float*)d_in[6];
    const float* gru_Whh  = (const float*)d_in[7];
    const float* gru_bih  = (const float*)d_in[8];
    const float* gru_bhh  = (const float*)d_in[9];
    const float* W1 = (const float*)d_in[10];
    const float* b1 = (const float*)d_in[11];
    const float* W2 = (const float*)d_in[12];
    const float* b2 = (const float*)d_in[13];
    const float* W3 = (const float*)d_in[14];
    const float* b3 = (const float*)d_in[15];
    float* out = (float*)d_out;

    const int N = in_sizes[0] / XDIM;       // 100000
    const int E = in_sizes[1] / 2;          // 1600000
    const int G = out_size;                 // 512
    const int* esrc = eidx;
    const int* edst = eidx + E;

    // ---- workspace layout ----
    __bf16* projF = (__bf16*)d_ws;                          // 3*4*512 = 6144
    __bf16* ggcF  = projF + 6144;                           // 8 * 4096 = 32768
    __bf16* wihF  = ggcF + 32768;                           // 4 * 12288 = 49152
    __bf16* whhF  = wihF + 49152;                           // 49152
    size_t fragBytes = (size_t)(6144 + 32768 + 49152 + 49152) * sizeof(__bf16); // 274432

    float* fbase  = (float*)((char*)d_ws + fragBytes);
    long   NH     = (long)N * HDIM;
    float* h      = fbase;
    float* m      = fbase + NH;
    float* agg    = fbase + 2 * NH;
    float* pooled = fbase + 3 * NH;                         // G*64
    float* counts = pooled + (long)G * HDIM;                // G

    const int ntiles = (N + 15) / 16;                       // 6250
    const int WPB    = 8;
    dim3 blk(256);
    dim3 gridTiles((ntiles + WPB - 1) / WPB);

    // ---- pack weights into WMMA B-fragment layout (bf16) ----
    {
        dim3 pb(256);
        pack_b_frags<<<(3 * 4 * 512 + 255) / 256, pb, 0, stream>>>(W_proj, XDIM, XDIM, 4, 3, 1, projF);
        for (int i = 0; i < NBLK * NLAY; ++i)
            pack_b_frags<<<(2 * 4 * 512 + 255) / 256, pb, 0, stream>>>(
                ggc_W + (long)i * HDIM * HDIM, HDIM, HDIM, 4, 2, 0, ggcF + (long)i * 4096);
        for (int b = 0; b < NBLK; ++b) {
            pack_b_frags<<<(2 * 12 * 512 + 255) / 256, pb, 0, stream>>>(
                gru_Wih + (long)b * 192 * HDIM, HDIM, HDIM, 12, 2, 1, wihF + (long)b * 12288);
            pack_b_frags<<<(2 * 12 * 512 + 255) / 256, pb, 0, stream>>>(
                gru_Whh + (long)b * 192 * HDIM, HDIM, HDIM, 12, 2, 1, whhF + (long)b * 12288);
        }
    }

    // ---- input projection: h = x @ Wproj.T + b  (K=79 padded to 96) ----
    gemm_bf16_wmma<4, 3, false><<<gridTiles, blk, 0, stream>>>(
        x, XDIM, XDIM, projF, b_proj, h, HDIM, nullptr, N, ntiles);

    // ---- 4 blocks x 2 GatedGraphConv layers ----
    for (int b = 0; b < NBLK; ++b) {
        for (int l = 0; l < NLAY; ++l) {
            // m = h @ W  (also zeroes agg for the scatter)
            gemm_bf16_wmma<4, 2, true><<<gridTiles, blk, 0, stream>>>(
                h, HDIM, HDIM, ggcF + (long)(b * NLAY + l) * 4096, nullptr, m, HDIM, agg, N, ntiles);
            long sthreads = (long)E * 32;
            scatter_edges<<<(int)((sthreads + 255) / 256), blk, 0, stream>>>(m, esrc, edst, agg, E);
            gru_fused<<<gridTiles, blk, 0, stream>>>(
                agg, h, wihF + (long)b * 12288, whhF + (long)b * 12288,
                gru_bih + (long)b * 192, gru_bhh + (long)b * 192,
                N, ntiles, (l == NLAY - 1) ? 1 : 0);
        }
    }

    // ---- mean pooling + MLP head ----
    long pz = (long)G * HDIM + G;
    zero_f32<<<(int)((pz + 255) / 256), blk, 0, stream>>>(pooled, pz);
    long pthreads = (long)N * 32;
    pool_scatter<<<(int)((pthreads + 255) / 256), blk, 0, stream>>>(h, batch, pooled, counts, N);
    head_kernel<<<(G + 255) / 256, blk, 0, stream>>>(pooled, counts, W1, b1, W2, b2, W3, b3, out, G);
}
